// PropeDotProductAttention_39986145525878
// MI455X (gfx1250) — matrix operand order repaired and checked by reading, not compiled, not run on hardware
//
#include <hip/hip_runtime.h>
#include <hip/hip_bf16.h>

#define HD 64
#define SS 4096   // S = C*PX*PY
#define NH 12     // heads
#define NC 4      // cameras
#define IWH 512.0f
#define BK 64     // keys per slab
#define PADH 72   // LDS row pitch in halves (144B = 36 dwords, conflict-free)

typedef _Float16 v16h __attribute__((ext_vector_type(16)));
typedef _Float16 v8h  __attribute__((ext_vector_type(8)));
typedef float    v8f  __attribute__((ext_vector_type(8)));

__device__ __forceinline__ v16h cat8(v8h a, v8h b) {
    return __builtin_shufflevector(a, b, 0,1,2,3,4,5,6,7,8,9,10,11,12,13,14,15);
}

// ---------------------------------------------------------------------------
// Kernel 0: per-camera matrices. P = liftK(Ksn) @ viewmat ; PT = P^T ;
// Pinv = invert_SE3(viewmat) @ liftK(invK(Ksn)).   One lane per camera.
// ---------------------------------------------------------------------------
__global__ void prope_mats(const float* __restrict__ VM, const float* __restrict__ Ks,
                           float* __restrict__ Pm, float* __restrict__ PTm,
                           float* __restrict__ Pinvm) {
    int c = threadIdx.x;
    if (c >= NC) return;
    const float* K = Ks + c * 9;
    const float* V = VM + c * 16;
    float kn00 = K[0] / IWH, kn11 = K[4] / IWH;
    float kn02 = K[2] / IWH - 0.5f, kn12 = K[5] / IWH - 0.5f;
    float L[16] = { kn00, 0.f, kn02, 0.f,
                    0.f, kn11, kn12, 0.f,
                    0.f, 0.f, 1.f, 0.f,
                    0.f, 0.f, 0.f, 1.f };
    float P[16];
    #pragma unroll
    for (int i = 0; i < 4; i++)
        #pragma unroll
        for (int j = 0; j < 4; j++) {
            float acc = 0.f;
            #pragma unroll
            for (int t = 0; t < 4; t++) acc += L[i*4+t] * V[t*4+j];
            P[i*4+j] = acc;
        }
    #pragma unroll
    for (int i = 0; i < 4; i++)
        #pragma unroll
        for (int j = 0; j < 4; j++) {
            Pm [c*16 + i*4 + j] = P[i*4+j];
            PTm[c*16 + i*4 + j] = P[j*4+i];
        }
    float Vi[16];
    #pragma unroll
    for (int i = 0; i < 16; i++) Vi[i] = 0.f;
    #pragma unroll
    for (int i = 0; i < 3; i++) {
        #pragma unroll
        for (int j = 0; j < 3; j++) Vi[i*4+j] = V[j*4+i];               // R^T
        Vi[i*4+3] = -(V[0*4+i]*V[3] + V[1*4+i]*V[7] + V[2*4+i]*V[11]);  // -R^T t
    }
    Vi[15] = 1.f;
    float Li[16] = { 1.f/kn00, 0.f, -kn02/kn00, 0.f,
                     0.f, 1.f/kn11, -kn12/kn11, 0.f,
                     0.f, 0.f, 1.f, 0.f,
                     0.f, 0.f, 0.f, 1.f };
    #pragma unroll
    for (int i = 0; i < 4; i++)
        #pragma unroll
        for (int j = 0; j < 4; j++) {
            float acc = 0.f;
            #pragma unroll
            for (int t = 0; t < 4; t++) acc += Vi[i*4+t] * Li[t*4+j];
            Pinvm[c*16 + i*4 + j] = acc;
        }
}

// ---------------------------------------------------------------------------
__device__ __forceinline__ void prope_fwd(const float* in, float* out, const float* M,
                                          const float* cxc, const float* cxs,
                                          const float* cyc, const float* cys) {
    #pragma unroll
    for (int b = 0; b < 8; b++) {
        const float* x4 = in + b*4;
        #pragma unroll
        for (int i = 0; i < 4; i++)
            out[b*4+i] = M[i*4+0]*x4[0] + M[i*4+1]*x4[1] + M[i*4+2]*x4[2] + M[i*4+3]*x4[3];
    }
    #pragma unroll
    for (int i = 0; i < 8; i++) {
        float xi = in[32+i], yi = in[40+i];
        out[32+i] =  cxc[i]*xi + cxs[i]*yi;
        out[40+i] = -cxs[i]*xi + cxc[i]*yi;
        xi = in[48+i]; yi = in[56+i];
        out[48+i] =  cyc[i]*xi + cys[i]*yi;
        out[56+i] = -cys[i]*xi + cyc[i]*yi;
    }
}

// ---------------------------------------------------------------------------
// Kernel 1: q,k,v (f32 [s][h][64]) -> f16 Qh/Kh [h][s][64], Vt [h][64][s].
// Q gets the 1/sqrt(HD) attention scale folded in.
// ---------------------------------------------------------------------------
__global__ __launch_bounds__(256)
void prope_pre(const float* __restrict__ q, const float* __restrict__ k,
               const float* __restrict__ v,
               const float* __restrict__ PTm, const float* __restrict__ Pinvm,
               _Float16* __restrict__ Qh, _Float16* __restrict__ Kh,
               _Float16* __restrict__ Vt) {
    int tid = blockIdx.x * blockDim.x + threadIdx.x;   // exactly SS*NH threads
    int s  = tid / NH;
    int hh = tid % NH;
    int cam = s >> 10;
    float px = (float)(s & 31);
    float py = (float)((s >> 5) & 31);

    float cxc[8], cxs[8], cyc[8], cys[8];
    #pragma unroll
    for (int i = 0; i < 8; i++) {
        float fr = __powf(100.0f, -(float)i / 8.0f);
        float ax = px * fr, ay = py * fr;
        cxc[i] = __cosf(ax); cxs[i] = __sinf(ax);
        cyc[i] = __cosf(ay); cys[i] = __sinf(ay);
    }
    float MT[16], MI[16];
    #pragma unroll
    for (int i = 0; i < 16; i++) { MT[i] = PTm[cam*16+i]; MI[i] = Pinvm[cam*16+i]; }

    float xin[64], xout[64];
    const size_t base = (size_t)tid * HD;

    // q -> Qh (P^T), scaled by 0.125
    #pragma unroll
    for (int d = 0; d < 64; d++) xin[d] = q[base + d];
    prope_fwd(xin, xout, MT, cxc, cxs, cyc, cys);
    {
        _Float16* dst = Qh + ((size_t)hh * SS + s) * HD;
        #pragma unroll
        for (int d = 0; d < 64; d++) dst[d] = (_Float16)(xout[d] * 0.125f);
    }
    // k -> Kh (P^-1)
    #pragma unroll
    for (int d = 0; d < 64; d++) xin[d] = k[base + d];
    prope_fwd(xin, xout, MI, cxc, cxs, cyc, cys);
    {
        _Float16* dst = Kh + ((size_t)hh * SS + s) * HD;
        #pragma unroll
        for (int d = 0; d < 64; d++) dst[d] = (_Float16)xout[d];
    }
    // v -> Vt transposed (P^-1)
    #pragma unroll
    for (int d = 0; d < 64; d++) xin[d] = v[base + d];
    prope_fwd(xin, xout, MI, cxc, cxs, cyc, cys);
    {
        _Float16* dst = Vt + (size_t)hh * HD * SS + s;
        #pragma unroll
        for (int d = 0; d < 64; d++) dst[(size_t)d * SS] = (_Float16)xout[d];
    }
}

// ---------------------------------------------------------------------------
// Kernel 2: flash attention. Block = 8 waves = 128 queries of one head.
// 64-key slabs staged in LDS with async global->LDS copies, double-buffered.
// Q@K^T and P@V on v_wmma_f32_16x16x32_f16; row sums of exp(P) via a
// ones-matrix WMMA (kills the sum butterfly).
// ---------------------------------------------------------------------------
__global__ __launch_bounds__(256)
void prope_attn(const _Float16* __restrict__ Qh, const _Float16* __restrict__ Kh,
                const _Float16* __restrict__ Vt, float* __restrict__ Out) {
    __shared__ __align__(16) _Float16 sK[2][64 * PADH];   // [key][feat]
    __shared__ __align__(16) _Float16 sV[2][64 * PADH];   // [feat d][key]
    __shared__ __align__(16) _Float16 psh[8][16 * PADH];  // per-wave P bounce

    const int tid  = threadIdx.x;
    const int lane = tid & 31;
    const int wib  = tid >> 5;
    const int head = blockIdx.x >> 5;          // 32 query blocks per head
    const int q0   = (blockIdx.x & 31) * 128 + wib * 16;
    const int hgrp = lane >> 4;
    const int l16  = lane & 15;

    const _Float16* Kbase = Kh + (size_t)head * SS * HD;
    const _Float16* Vbase = Vt + (size_t)head * HD * SS;

    // async-copy slice for this thread: rows r0 and r0+32, 16B chunk c0
    const int r0 = tid >> 3;
    const int c0 = (tid & 7) * 8;   // halves
    auto issue_slab = [&](int kt, int b) {
        const _Float16* gk0 = Kbase + (size_t)(kt + r0) * HD + c0;
        const _Float16* gk1 = Kbase + (size_t)(kt + r0 + 32) * HD + c0;
        const _Float16* gv0 = Vbase + (size_t)r0 * SS + kt + c0;
        const _Float16* gv1 = Vbase + (size_t)(r0 + 32) * SS + kt + c0;
        uint32_t lk0 = (uint32_t)(uintptr_t)&sK[b][r0 * PADH + c0];
        uint32_t lk1 = (uint32_t)(uintptr_t)&sK[b][(r0 + 32) * PADH + c0];
        uint32_t lv0 = (uint32_t)(uintptr_t)&sV[b][r0 * PADH + c0];
        uint32_t lv1 = (uint32_t)(uintptr_t)&sV[b][(r0 + 32) * PADH + c0];
        asm volatile("global_load_async_to_lds_b128 %0, %1, off" :: "v"(lk0), "v"(gk0) : "memory");
        asm volatile("global_load_async_to_lds_b128 %0, %1, off" :: "v"(lk1), "v"(gk1) : "memory");
        asm volatile("global_load_async_to_lds_b128 %0, %1, off" :: "v"(lv0), "v"(gv0) : "memory");
        asm volatile("global_load_async_to_lds_b128 %0, %1, off" :: "v"(lv1), "v"(gv1) : "memory");
    };

    // Q tile in WMMA A-layout (two 16x32 f16 operands covering HD=64)
    const _Float16* Qrow = Qh + ((size_t)head * SS + (q0 + l16)) * HD;
    v16h qa0 = cat8(*(const v8h*)(Qrow +      hgrp * 8),
                    *(const v8h*)(Qrow + 16 + hgrp * 8));
    v16h qa1 = cat8(*(const v8h*)(Qrow + 32 + hgrp * 8),
                    *(const v8h*)(Qrow + 48 + hgrp * 8));

    v16h onesb;
    #pragma unroll
    for (int e = 0; e < 16; e++) onesb[e] = (_Float16)1.0f;

    float m_run[8], l_run[8], alpha[8];
    v8f o[4] = { v8f{}, v8f{}, v8f{}, v8f{} };
    #pragma unroll
    for (int r = 0; r < 8; r++) { m_run[r] = -1e30f; l_run[r] = 0.0f; }

    issue_slab(0, 0);
    asm volatile("s_wait_asynccnt 0x0" ::: "memory");
    __syncthreads();

    int buf = 0;
    for (int kt = 0; kt < SS; kt += BK, buf ^= 1) {
        if (kt + BK < SS) issue_slab(kt + BK, buf ^ 1);   // prefetch next slab

        // --- scores: four 16-key column tiles, each accumulated over HD ---
        v8f sc[4];
        #pragma unroll
        for (int t = 0; t < 4; t++) {
            const _Float16* kr = &sK[buf][(t * 16 + l16) * PADH + hgrp * 8];
            v16h kb0 = cat8(*(const v8h*)(kr),      *(const v8h*)(kr + 16));
            v16h kb1 = cat8(*(const v8h*)(kr + 32), *(const v8h*)(kr + 48));
            v8f s = {};
            s = __builtin_amdgcn_wmma_f32_16x16x32_f16(false, qa0, false, kb0, (short)0, s, false, false);
            s = __builtin_amdgcn_wmma_f32_16x16x32_f16(false, qa1, false, kb1, (short)0, s, false, false);
            sc[t] = s;
        }

        // --- online softmax: row max butterfly + exp, P -> LDS ---
        _Float16* prow = &psh[wib][0];
        #pragma unroll
        for (int r = 0; r < 8; r++) {
            float a0 = sc[0][r], a1 = sc[1][r], a2 = sc[2][r], a3 = sc[3][r];
            float mx = fmaxf(fmaxf(a0, a1), fmaxf(a2, a3));
            mx = fmaxf(mx, __shfl_xor(mx, 1, 32));
            mx = fmaxf(mx, __shfl_xor(mx, 2, 32));
            mx = fmaxf(mx, __shfl_xor(mx, 4, 32));
            mx = fmaxf(mx, __shfl_xor(mx, 8, 32));
            float mn = fmaxf(m_run[r], mx);
            alpha[r] = __expf(m_run[r] - mn);
            m_run[r] = mn;
            float p0 = __expf(a0 - mn), p1 = __expf(a1 - mn);
            float p2 = __expf(a2 - mn), p3 = __expf(a3 - mn);
            o[0][r] *= alpha[r]; o[1][r] *= alpha[r];
            o[2][r] *= alpha[r]; o[3][r] *= alpha[r];
            int row = (r + hgrp * 8) * PADH;
            prow[row + l16]      = (_Float16)p0;
            prow[row + 16 + l16] = (_Float16)p1;
            prow[row + 32 + l16] = (_Float16)p2;
            prow[row + 48 + l16] = (_Float16)p3;
        }
        asm volatile("" ::: "memory");   // DS in-order per wave; keep stores first

        // --- reload P tile in A-layout (16x64 -> two 16x32 operands) ---
        const _Float16* par = &psh[wib][l16 * PADH + hgrp * 8];
        v16h pa0 = cat8(*(const v8h*)(par),      *(const v8h*)(par + 16));
        v16h pa1 = cat8(*(const v8h*)(par + 32), *(const v8h*)(par + 48));

        // --- row sums of P via ones-matrix WMMA (replaces sum butterfly) ---
        v8f ls = {};
        ls = __builtin_amdgcn_wmma_f32_16x16x32_f16(false, pa0, false, onesb, (short)0, ls, false, false);
        ls = __builtin_amdgcn_wmma_f32_16x16x32_f16(false, pa1, false, onesb, (short)0, ls, false, false);
        #pragma unroll
        for (int r = 0; r < 8; r++) l_run[r] = l_run[r] * alpha[r] + ls[r];

        // --- O += P @ V (four 16x16 d-chunks, 64 keys) ---
        #pragma unroll
        for (int ch = 0; ch < 4; ch++) {
            const _Float16* vr = &sV[buf][(ch * 16 + l16) * PADH + hgrp * 8];
            v16h vb0 = cat8(*(const v8h*)(vr),      *(const v8h*)(vr + 16));
            v16h vb1 = cat8(*(const v8h*)(vr + 32), *(const v8h*)(vr + 48));
            o[ch] = __builtin_amdgcn_wmma_f32_16x16x32_f16(false, pa0, false, vb0, (short)0, o[ch], false, false);
            o[ch] = __builtin_amdgcn_wmma_f32_16x16x32_f16(false, pa1, false, vb1, (short)0, o[ch], false, false);
        }

        asm volatile("s_wait_asynccnt 0x0" ::: "memory");  // prefetch landed
        __syncthreads();                                   // all waves done with buf
    }

    // --- normalize and store raw attention output: Out[s][h][d] (f32) ---
    #pragma unroll
    for (int r = 0; r < 8; r++) {
        float inv = 1.0f / l_run[r];
        int qrow = q0 + r + hgrp * 8;
        float* op = Out + ((size_t)qrow * NH + head) * HD;
        #pragma unroll
        for (int ch = 0; ch < 4; ch++)
            op[ch * 16 + l16] = o[ch][r] * inv;
    }
}

// ---------------------------------------------------------------------------
// Kernel 3: in-place inverse PRoPE on d_out: _proj with P, inverse RoPE.
// ---------------------------------------------------------------------------
__global__ __launch_bounds__(256)
void prope_post(const float* __restrict__ Pm, float* __restrict__ Out) {
    int tid = blockIdx.x * blockDim.x + threadIdx.x;
    int s  = tid / NH;
    int cam = s >> 10;
    float px = (float)(s & 31);
    float py = (float)((s >> 5) & 31);

    float cxc[8], cxs[8], cyc[8], cys[8];
    #pragma unroll
    for (int i = 0; i < 8; i++) {
        float fr = __powf(100.0f, -(float)i / 8.0f);
        float ax = px * fr, ay = py * fr;
        cxc[i] = __cosf(ax); cxs[i] = __sinf(ax);
        cyc[i] = __cosf(ay); cys[i] = __sinf(ay);
    }
    float M[16];
    #pragma unroll
    for (int i = 0; i < 16; i++) M[i] = Pm[cam*16 + i];

    float* base = Out + (size_t)tid * HD;
    float xin[64], xout[64];
    #pragma unroll
    for (int d = 0; d < 64; d++) xin[d] = base[d];

    #pragma unroll
    for (int b = 0; b < 8; b++) {
        const float* x4 = xin + b*4;
        #pragma unroll
        for (int i = 0; i < 4; i++)
            xout[b*4+i] = M[i*4+0]*x4[0] + M[i*4+1]*x4[1] + M[i*4+2]*x4[2] + M[i*4+3]*x4[3];
    }
    #pragma unroll
    for (int i = 0; i < 8; i++) {       // inverse RoPE
        float xi = xin[32+i], yi = xin[40+i];
        xout[32+i] = cxc[i]*xi - cxs[i]*yi;
        xout[40+i] = cxs[i]*xi + cxc[i]*yi;
        xi = xin[48+i]; yi = xin[56+i];
        xout[48+i] = cyc[i]*xi - cys[i]*yi;
        xout[56+i] = cys[i]*xi + cyc[i]*yi;
    }
    #pragma unroll
    for (int d = 0; d < 64; d++) base[d] = xout[d];
}

// ---------------------------------------------------------------------------
extern "C" void kernel_launch(void* const* d_in, const int* in_sizes, int n_in,
                              void* d_out, int out_size, void* d_ws, size_t ws_size,
                              hipStream_t stream) {
    const float* q  = (const float*)d_in[0];
    const float* k  = (const float*)d_in[1];
    const float* v  = (const float*)d_in[2];
    const float* vm = (const float*)d_in[3];
    const float* Ks = (const float*)d_in[4];
    float* out = (float*)d_out;

    char* ws = (char*)d_ws;
    float* Pm    = (float*)ws;          // 64 floats
    float* PTm   = Pm + 64;
    float* Pinvm = Pm + 128;
    _Float16* Qh = (_Float16*)(ws + 1024);
    _Float16* Kh = Qh + (size_t)NH * SS * HD;
    _Float16* Vt = Kh + (size_t)NH * SS * HD;

    prope_mats<<<1, 32, 0, stream>>>(vm, Ks, Pm, PTm, Pinvm);
    prope_pre<<<(SS * NH) / 256, 256, 0, stream>>>(q, k, v, PTm, Pinvm, Qh, Kh, Vt);
    prope_attn<<<NH * (SS / 128), 256, 0, stream>>>(Qh, Kh, Vt, out);
    prope_post<<<(SS * NH) / 256, 256, 0, stream>>>(Pm, out);
}
